// Dynamic_Fusion_34316788695409
// MI455X (gfx1250) — matrix operand order repaired and checked by reading, not compiled, not run on hardware
//
#include <hip/hip_runtime.h>

typedef __attribute__((ext_vector_type(16))) _Float16 v16h;
typedef __attribute__((ext_vector_type(8)))  _Float16 v8h;
typedef __attribute__((ext_vector_type(8)))  float    v8f;
typedef __attribute__((ext_vector_type(4)))  unsigned int v4u;
typedef __attribute__((ext_vector_type(8)))  int      v8i;
typedef __attribute__((ext_vector_type(4)))  int      v4i;

#define NB   32
#define NT   512
#define NF   128
#define NHID 256
#define NTW  1024
#define NOUT 4
#define XS_W 24   // staged time window: 16 output cols + up to 8 of causal history

#if __has_builtin(__builtin_amdgcn_tensor_load_to_lds)
  #define TDM_AVAILABLE 1
  #if __has_include(<hip/amd_detail/amd_gfx1250_TDM.h>)
    #define TDM_SIX_ARGS 1
  #else
    #define TDM_SIX_ARGS 0
  #endif
#else
  #define TDM_AVAILABLE 0
#endif

// ---------- WMMA fragment helpers (CDNA5 16x16x32 f16 layouts, wave32) ----------
__device__ __forceinline__ v16h cat16(v8h a, v8h b) {
  return __builtin_shufflevector(a, b, 0,1,2,3,4,5,6,7,8,9,10,11,12,13,14,15);
}
// A-matrix 16x32 f16: lane holds row m=lane&15; element j -> k = 16*(lane>=16) + j
__device__ __forceinline__ v16h frag_a16(const _Float16* rowk0, int g) {
  v8h c0 = *(const v8h*)(rowk0 + 16 * g);
  v8h c1 = *(const v8h*)(rowk0 + 16 * g + 8);
  return cat16(c0, c1);
}
// B-matrix 32x16 f16: lane holds col n=lane&15; element j -> k = 16*(lane>=16) + j
__device__ __forceinline__ v16h frag_b16(const _Float16* rowk0, int g) {
  v8h c0 = *(const v8h*)(rowk0 + 16 * g);
  v8h c1 = *(const v8h*)(rowk0 + 16 * g + 8);
  return cat16(c0, c1);
}
__device__ __forceinline__ v8f wmma16(v16h a, v16h b, v8f c) {
  return __builtin_amdgcn_wmma_f32_16x16x32_f16(false, a, false, b, (short)0, c, false, false);
}

// ---------- Tensor Data Mover: 2D tile (tile_t x tile_c), element = 2 bytes ----------
// D# per CDNA5 ISA ch.8: group0 = {count/type, lds_addr, global_addr}, group1 = dims/strides.
__device__ __forceinline__ void tdm_load_2d_f16(const _Float16* gsrc, unsigned lds_byte_off,
                                                unsigned tile_t, unsigned tile_c,
                                                unsigned tensor_t, unsigned tensor_c,
                                                unsigned stride_t_elems) {
#if TDM_AVAILABLE
  unsigned long long ga = (unsigned long long)(size_t)gsrc;
  v4u g0;
  g0.x = 1u;                                                 // count=1, user descriptor
  g0.y = lds_byte_off;                                       // lds_addr (bytes)
  g0.z = (unsigned)(ga & 0xffffffffu);                       // global_addr[31:0]
  g0.w = (unsigned)((ga >> 32) & 0x01ffffffu) | (2u << 30);  // global_addr[56:32], type=2
  v8i g1;
  g1[0] = (int)(1u << 16);                                   // workgroup_mask=0, data_size=1 (2B)
  g1[1] = (int)((tensor_t & 0xffffu) << 16);                 // tensor_dim0[15:0]
  g1[2] = (int)(((tensor_t >> 16) & 0xffffu) | ((tensor_c & 0xffffu) << 16));
  g1[3] = (int)(((tensor_c >> 16) & 0xffffu) | ((tile_t & 0xffffu) << 16));
  g1[4] = (int)(tile_c & 0xffffu);                           // tile_dim1; tile_dim2=0
  g1[5] = (int)stride_t_elems;                               // tensor_dim0_stride[31:0]
  g1[6] = 0;                                                 // stride0[47:32], stride1[15:0]
  g1[7] = 0;
  v4i z4 = {0, 0, 0, 0};
  #if TDM_SIX_ARGS
  v8i z8 = {0, 0, 0, 0, 0, 0, 0, 0};
  __builtin_amdgcn_tensor_load_to_lds(g0, g1, z4, z4, z8, 0);
  #else
  __builtin_amdgcn_tensor_load_to_lds(g0, g1, z4, z4, 0);
  #endif
#else
  (void)gsrc; (void)lds_byte_off; (void)tile_t; (void)tile_c;
  (void)tensor_t; (void)tensor_c; (void)stride_t_elems;
#endif
}

__device__ __forceinline__ void wait_tensor0() {
#if __has_builtin(__builtin_amdgcn_s_wait_tensorcnt)
  __builtin_amdgcn_s_wait_tensorcnt(0);
#endif
  __asm__ volatile("" ::: "memory");
}

// ---------- small elementwise kernels ----------
__global__ void cvt_f32_f16_kernel(const float* __restrict__ s, _Float16* __restrict__ d, int n) {
  int i = blockIdx.x * blockDim.x + threadIdx.x;
  if (i < n) d[i] = (_Float16)s[i];
}

// conv_w [O][I][3] f32 -> [3][O][I] f16 (per-tap contiguous GEMM operands)
__global__ void repack_conv_kernel(const float* __restrict__ s, _Float16* __restrict__ d, int O, int I) {
  int i = blockIdx.x * blockDim.x + threadIdx.x;
  int n = O * I * 3;
  if (i >= n) return;
  int o = i / (I * 3);
  int rem = i % (I * 3);
  int c = rem / 3;
  int k = rem % 3;
  d[(size_t)k * O * I + (size_t)o * I + c] = (_Float16)s[i];
}

// feature [B,F,1,T] f32 -> [B,T,F] f32 + f16
__global__ void prep_feat_kernel(const float* __restrict__ src, float* __restrict__ df, _Float16* __restrict__ dh) {
  int i = blockIdx.x * blockDim.x + threadIdx.x;
  if (i >= NB * NT * NF) return;
  int b = i / (NT * NF);
  int r = i % (NT * NF);
  int t = r / NF;
  int f = r % NF;
  float v = src[((size_t)b * NF + f) * NT + t];
  df[i] = v;
  dh[i] = (_Float16)v;
}

// ---------- projection GEMM: Y[m,n] = sum_k X[m,k] * W[n,k] + bias[n] ----------
__global__ void gemm_proj_kernel(const _Float16* __restrict__ X, const _Float16* __restrict__ W,
                                 const float* __restrict__ bias, _Float16* __restrict__ Y) {
  int lane = threadIdx.x;
  int g = lane >> 4, nn = lane & 15;
  int m0 = blockIdx.x * 16, n0 = blockIdx.y * 16;
  v8f acc = {};
  for (int k0 = 0; k0 < NF; k0 += 32) {
    v16h a = frag_a16(X + (size_t)(m0 + nn) * NF + k0, g);
    v16h b = frag_b16(W + (size_t)(n0 + nn) * NF + k0, g);
    acc = wmma16(a, b, acc);
  }
  float bv = bias[n0 + nn];
#pragma unroll
  for (int r = 0; r < 8; ++r) {
    int m = m0 + r + 8 * g;
    Y[(size_t)m * NF + n0 + nn] = (_Float16)(acc[r] + bv);
  }
}

// ---------- fused attention: scores->softmax->PV, in-place gated residual on KV-source ----------
// Phase-3 V slab is staged column-tile-at-a-time into LDS by the Tensor Data Mover.
__global__ void attn_kernel(const _Float16* __restrict__ Qh, const _Float16* __restrict__ Kh,
                            const _Float16* __restrict__ Vh, float* __restrict__ tgtF,
                            _Float16* __restrict__ tgtH) {
  __shared__ __align__(16) float    sc[16][NT];
  __shared__ __align__(16) _Float16 ph[16][NT];
#if TDM_AVAILABLE
  __shared__ __align__(16) _Float16 vs[NT * 16];   // V[:, n0:n0+16] slab (16 KB)
#endif
  int lane = threadIdx.x;
  int g = lane >> 4, nn = lane & 15;
  int b = blockIdx.y;
  int q0 = blockIdx.x * 16;
  const _Float16* Qb = Qh + (size_t)b * NT * NF;
  const _Float16* Kb = Kh + (size_t)b * NT * NF;
  const _Float16* Vb = Vh + (size_t)b * NT * NF;
  const float scale = 0.08838834764831845f; // 1/sqrt(128)

  // Phase 1: score tiles via WMMA -> LDS
  for (int s0 = 0; s0 < NT; s0 += 16) {
    v8f acc = {};
    for (int k0 = 0; k0 < NF; k0 += 32) {
      v16h a = frag_a16(Qb + (size_t)(q0 + nn) * NF + k0, g);
      v16h bb = frag_b16(Kb + (size_t)(s0 + nn) * NF + k0, g);
      acc = wmma16(a, bb, acc);
    }
#pragma unroll
    for (int r = 0; r < 8; ++r) sc[r + 8 * g][s0 + nn] = acc[r] * scale;
  }
  __syncthreads();

  // Phase 2: softmax per row (2 lanes per row, cross-half combine via shfl_xor 16)
  {
    int row = lane & 15, half = lane >> 4;
    int s_beg = half * (NT / 2), s_end = s_beg + (NT / 2);
    float mx = -3.0e38f;
    for (int s = s_beg; s < s_end; ++s) mx = fmaxf(mx, sc[row][s]);
    mx = fmaxf(mx, __shfl_xor(mx, 16));
    float sum = 0.0f;
    for (int s = s_beg; s < s_end; ++s) {
      float e = __expf(sc[row][s] - mx);
      sc[row][s] = e;
      sum += e;
    }
    sum += __shfl_xor(sum, 16);
    float inv = 1.0f / sum;
    for (int s = s_beg; s < s_end; ++s) ph[row][s] = (_Float16)(sc[row][s] * inv);
  }
  __syncthreads();

  // Phase 3: out = P @ V via WMMA, then gated residual update in place
  for (int n0 = 0; n0 < NF; n0 += 16) {
#if TDM_AVAILABLE
    // DMA V[s=0..511][n0..n0+15] -> vs[s*16 + n]
    tdm_load_2d_f16(Vb + n0, (unsigned)(size_t)(&vs[0]),
                    16u, NT, (unsigned)(NF - n0), NT, NF);
    wait_tensor0();
    __syncthreads();
#endif
    v8f acc = {};
    for (int s0 = 0; s0 < NT; s0 += 32) {
      v16h a = frag_a16(&ph[nn][s0], g);
      v16h bb;
#pragma unroll
      for (int j = 0; j < 16; ++j) {
#if TDM_AVAILABLE
        bb[j] = vs[(s0 + 16 * g + j) * 16 + nn];
#else
        bb[j] = Vb[(size_t)(s0 + 16 * g + j) * NF + n0 + nn];
#endif
      }
      acc = wmma16(a, bb, acc);
    }
#pragma unroll
    for (int r = 0; r < 8; ++r) {
      int m = q0 + r + 8 * g;
      size_t idx = ((size_t)b * NT + m) * NF + n0 + nn;
      float old = tgtF[idx];
      float nv = acc[r] * old + old;
      tgtF[idx] = nv;
      tgtH[idx] = (_Float16)nv;
    }
#if TDM_AVAILABLE
    __syncthreads();   // protect vs before next tile's DMA overwrites it
#endif
  }
}

// ---------- concat (PAD_LEN==T): x[b,c,t] = t<512 ? f1[b,t,c] : f2[b,t-512,c] ----------
__global__ void concat_kernel(const _Float16* __restrict__ f1h, const _Float16* __restrict__ f2h,
                              _Float16* __restrict__ x0h) {
  int i = blockIdx.x * blockDim.x + threadIdx.x;
  if (i >= NB * NF * NTW) return;
  int b = i / (NF * NTW);
  int r = i % (NF * NTW);
  int c = r / NTW;
  int t = r % NTW;
  _Float16 v;
  if (t < NT) v = f1h[((size_t)b * NT + t) * NF + c];
  else        v = f2h[((size_t)b * NT + (t - NT)) * NF + c];
  x0h[i] = v;
}

// ---------- TCN layer with TDM-staged input slab ----------
// Block = 1 wave, handles (b, 16-wide t-tile) for ALL output channels, reusing the
// [Cin x 24] X slab staged in LDS by the Tensor Data Mover.
// out = relu(sum_k Wk @ shift(X,(2-k)*d) + cb) + (Wr @ X + rb)
__global__ void tcn_layer_kernel(const _Float16* __restrict__ Xh, int Cin,
                                 const _Float16* __restrict__ Wc, const float* __restrict__ cb,
                                 const _Float16* __restrict__ Wr, const float* __restrict__ rb,
                                 int dil, _Float16* __restrict__ Yh, float* __restrict__ Partial) {
  __shared__ __align__(16) _Float16 xs[NHID * XS_W];   // up to 12 KB slab
  __shared__ float tile[16][17];
  int lane = threadIdx.x;
  int g = lane >> 4, nn = lane & 15;
  int t0 = blockIdx.x * 16, b = blockIdx.y;
  const _Float16* Xb = Xh + (size_t)b * Cin * NTW;
  int tstart = (t0 >= 8) ? (t0 - 8) : 0;

#if TDM_AVAILABLE
  // DMA X[c=0..Cin-1][tstart..tstart+23] -> xs[c*24 + (t-tstart)]
  tdm_load_2d_f16(Xb + tstart, (unsigned)(size_t)(&xs[0]),
                  XS_W, (unsigned)Cin, (unsigned)(NTW - tstart), (unsigned)Cin, NTW);
  wait_tensor0();
#else
  for (int i = lane; i < Cin * XS_W; i += 32) {
    int c = i / XS_W, tt = i % XS_W;
    xs[i] = Xb[(size_t)c * NTW + tstart + tt];
  }
#endif
  __syncthreads();

  int tcol = t0 + nn;
  for (int o0 = 0; o0 < NHID; o0 += 16) {
    v8f accC = {}, accR = {};
    for (int k = 0; k < 3; ++k) {
      int ts = tcol - (2 - k) * dil;   // absolute time of this tap
      int tl = ts - tstart;            // LDS column
      const _Float16* Wk = Wc + (size_t)k * NHID * Cin;
      for (int c0 = 0; c0 < Cin; c0 += 32) {
        v16h a = frag_a16(Wk + (size_t)(o0 + nn) * Cin + c0, g);
        v16h bb;
#pragma unroll
        for (int j = 0; j < 16; ++j) {
          int c = c0 + 16 * g + j;
          bb[j] = (ts >= 0) ? xs[c * XS_W + tl] : (_Float16)0.0f;
        }
        accC = wmma16(a, bb, accC);
      }
    }
    {
      int tl = tcol - tstart;
      for (int c0 = 0; c0 < Cin; c0 += 32) {
        v16h a = frag_a16(Wr + (size_t)(o0 + nn) * Cin + c0, g);
        v16h bb;
#pragma unroll
        for (int j = 0; j < 16; ++j) bb[j] = xs[(c0 + 16 * g + j) * XS_W + tl];
        accR = wmma16(a, bb, accR);
      }
    }
#pragma unroll
    for (int r = 0; r < 8; ++r) {
      int o = o0 + r + 8 * g;
      float v = fmaxf(accC[r] + cb[o], 0.0f) + accR[r] + rb[o];
      if (Yh) Yh[((size_t)b * NHID + o) * NTW + tcol] = (_Float16)v;
      tile[r + 8 * g][nn] = v;
    }
    if (Partial) {
      __syncthreads();
      if (lane < 16) {
        float s = 0.0f;
        for (int n = 0; n < 16; ++n) s += tile[lane][n];
        Partial[((size_t)b * NHID + o0 + lane) * (NTW / 16) + blockIdx.x] = s;
      }
      __syncthreads();
    }
  }
}

// ---------- pooling + FC ----------
__global__ void pool_kernel(const float* __restrict__ Partial, float* __restrict__ pooled) {
  int i = blockIdx.x * blockDim.x + threadIdx.x;
  if (i >= NB * NHID) return;
  float s = 0.0f;
  for (int tt = 0; tt < NTW / 16; ++tt) s += Partial[(size_t)i * (NTW / 16) + tt];
  pooled[i] = s * (1.0f / (float)NTW);
}

__global__ void fc_kernel(const float* __restrict__ pooled, const float* __restrict__ fcw,
                          const float* __restrict__ fcb, float* __restrict__ out) {
  int i = threadIdx.x;
  if (i >= NB * NOUT) return;
  int b = i / NOUT, o = i % NOUT;
  float acc = fcb[o];
  for (int h = 0; h < NHID; ++h) acc += pooled[b * NHID + h] * fcw[o * NHID + h];
  out[i] = acc;
}

// ---------- MMD: squared norms, WMMA Gram, exp reduce ----------
__global__ void sqnorm_kernel(const float* __restrict__ f, float* __restrict__ sx) {
  int i = blockIdx.x * blockDim.x + threadIdx.x;
  if (i >= NB * NT) return;
  float s = 0.0f;
  for (int k = 0; k < NF; ++k) {
    float v = f[(size_t)i * NF + k];
    s += v * v;
  }
  sx[i] = s;
}

// G[t][i][j] = dot(Xa[i,t,:], Xb[j,t,:]); grid (T, 2, 2), one 16x16 tile per block
__global__ void gram_kernel(const _Float16* __restrict__ Xa, const _Float16* __restrict__ Xb,
                            float* __restrict__ G) {
  int lane = threadIdx.x;
  int g = lane >> 4, nn = lane & 15;
  int t = blockIdx.x;
  int i0 = blockIdx.y * 16, j0 = blockIdx.z * 16;
  v8f acc = {};
  for (int k0 = 0; k0 < NF; k0 += 32) {
    v16h a = frag_a16(Xa + ((size_t)(i0 + nn) * NT + t) * NF + k0, g);
    v16h b = frag_b16(Xb + ((size_t)(j0 + nn) * NT + t) * NF + k0, g);
    acc = wmma16(a, b, acc);
  }
#pragma unroll
  for (int r = 0; r < 8; ++r)
    G[(size_t)t * (NB * NB) + (i0 + r + 8 * g) * NB + j0 + nn] = acc[r];
}

__global__ void mmd_partial_kernel(const float* __restrict__ Gxx, const float* __restrict__ Gyy,
                                   const float* __restrict__ Gxy, const float* __restrict__ sx1,
                                   const float* __restrict__ sx2, float* __restrict__ partial) {
  __shared__ float red[256];
  int t = blockIdx.x;
  int tid = threadIdx.x;
  float local = 0.0f;
  for (int p = tid; p < NB * NB; p += 256) {
    int i = p >> 5, j = p & 31;
    float s1i = sx1[i * NT + t], s1j = sx1[j * NT + t];
    float s2i = sx2[i * NT + t], s2j = sx2[j * NT + t];
    size_t base = (size_t)t * (NB * NB) + p;
    float dxx = s1i + s1j - 2.0f * Gxx[base];
    float dyy = s2i + s2j - 2.0f * Gyy[base];
    float dxy = s1i + s2j - 2.0f * Gxy[base];
    local += __expf(-0.5f * dxx) + __expf(-0.5f * dyy) - 2.0f * __expf(-0.5f * dxy);
  }
  red[tid] = local;
  __syncthreads();
  for (int s = 128; s > 0; s >>= 1) {
    if (tid < s) red[tid] += red[tid + s];
    __syncthreads();
  }
  if (tid == 0) partial[t] = red[0];
}

__global__ void mmd_final_kernel(const float* __restrict__ partial, float* __restrict__ out_loss) {
  __shared__ float red[256];
  int tid = threadIdx.x;
  red[tid] = partial[tid] + partial[tid + 256];
  __syncthreads();
  for (int s = 128; s > 0; s >>= 1) {
    if (tid < s) red[tid] += red[tid + s];
    __syncthreads();
  }
  if (tid == 0) out_loss[0] = red[0] * (1.0f / ((float)NB * (float)NB * (float)NT));
}

// ============================ host launcher ============================
extern "C" void kernel_launch(void* const* d_in, const int* in_sizes, int n_in,
                              void* d_out, int out_size, void* d_ws, size_t ws_size,
                              hipStream_t stream) {
  (void)in_sizes; (void)n_in; (void)out_size; (void)ws_size;
  const float* feature1 = (const float*)d_in[0];
  const float* feature2 = (const float*)d_in[1];
  const float* wq_a = (const float*)d_in[2];  const float* bq_a = (const float*)d_in[3];
  const float* wk_a = (const float*)d_in[4];  const float* bk_a = (const float*)d_in[5];
  const float* wv_a = (const float*)d_in[6];  const float* bv_a = (const float*)d_in[7];
  const float* wq_b = (const float*)d_in[8];  const float* bq_b = (const float*)d_in[9];
  const float* wk_b = (const float*)d_in[10]; const float* bk_b = (const float*)d_in[11];
  const float* wv_b = (const float*)d_in[12]; const float* bv_b = (const float*)d_in[13];
  const float* conv_w[3] = {(const float*)d_in[14], (const float*)d_in[18], (const float*)d_in[22]};
  const float* conv_b[3] = {(const float*)d_in[15], (const float*)d_in[19], (const float*)d_in[23]};
  const float* res_w[3]  = {(const float*)d_in[16], (const float*)d_in[20], (const float*)d_in[24]};
  const float* res_b[3]  = {(const float*)d_in[17], (const float*)d_in[21], (const float*)d_in[25]};
  const float* fc_w = (const float*)d_in[26];
  const float* fc_b = (const float*)d_in[27];
  float* out = (float*)d_out;

  // workspace layout
  char* ws = (char*)d_ws;
  size_t cur = 0;
  auto alloc = [&](size_t bytes) { size_t o = cur; cur = (cur + bytes + 255) & ~(size_t)255; return o; };
  const size_t nBTF = (size_t)NB * NT * NF;

  float*    f1f = (float*)(ws + alloc(nBTF * 4));
  float*    f2f = (float*)(ws + alloc(nBTF * 4));
  _Float16* f1h = (_Float16*)(ws + alloc(nBTF * 2));
  _Float16* f2h = (_Float16*)(ws + alloc(nBTF * 2));
  _Float16* Qh  = (_Float16*)(ws + alloc(nBTF * 2));
  _Float16* Kh  = (_Float16*)(ws + alloc(nBTF * 2));
  _Float16* Vh  = (_Float16*)(ws + alloc(nBTF * 2));
  _Float16* wh[6];
  const float* wsrc[6] = {wq_a, wk_a, wv_a, wq_b, wk_b, wv_b};
  for (int i = 0; i < 6; ++i) wh[i] = (_Float16*)(ws + alloc((size_t)NF * NF * 2));
  int cin_l[3] = {NF, NHID, NHID};
  _Float16* cwh[3]; _Float16* rwh[3];
  for (int l = 0; l < 3; ++l) {
    cwh[l] = (_Float16*)(ws + alloc((size_t)3 * NHID * cin_l[l] * 2));
    rwh[l] = (_Float16*)(ws + alloc((size_t)NHID * cin_l[l] * 2));
  }
  _Float16* x0h = (_Float16*)(ws + alloc((size_t)NB * NF * NTW * 2));
  _Float16* x1h = (_Float16*)(ws + alloc((size_t)NB * NHID * NTW * 2));
  _Float16* x2h = (_Float16*)(ws + alloc((size_t)NB * NHID * NTW * 2));
  float* Partial = (float*)(ws + alloc((size_t)NB * NHID * (NTW / 16) * 4));
  float* pooled  = (float*)(ws + alloc((size_t)NB * NHID * 4));
  float* sx1 = (float*)(ws + alloc((size_t)NB * NT * 4));
  float* sx2 = (float*)(ws + alloc((size_t)NB * NT * 4));
  float* Gxx = (float*)(ws + alloc((size_t)NT * NB * NB * 4));
  float* Gyy = (float*)(ws + alloc((size_t)NT * NB * NB * 4));
  float* Gxy = (float*)(ws + alloc((size_t)NT * NB * NB * 4));
  float* mmdp = (float*)(ws + alloc((size_t)NT * 4));

  auto blocks = [](size_t n, int bs) { return (unsigned)((n + bs - 1) / bs); };

  // weight conversion
  for (int i = 0; i < 6; ++i)
    cvt_f32_f16_kernel<<<blocks((size_t)NF * NF, 256), 256, 0, stream>>>(wsrc[i], wh[i], NF * NF);
  for (int l = 0; l < 3; ++l) {
    repack_conv_kernel<<<blocks((size_t)NHID * cin_l[l] * 3, 256), 256, 0, stream>>>(conv_w[l], cwh[l], NHID, cin_l[l]);
    cvt_f32_f16_kernel<<<blocks((size_t)NHID * cin_l[l], 256), 256, 0, stream>>>(res_w[l], rwh[l], NHID * cin_l[l]);
  }

  // features [B,F,1,T] -> [B,T,F]
  prep_feat_kernel<<<blocks(nBTF, 256), 256, 0, stream>>>(feature1, f1f, f1h);
  prep_feat_kernel<<<blocks(nBTF, 256), 256, 0, stream>>>(feature2, f2f, f2h);

  dim3 gProj(NB * NT / 16, NF / 16);
  // branch A: Q from f2, K/V from f1, gated update on f1
  gemm_proj_kernel<<<gProj, 32, 0, stream>>>(f2h, wh[0], bq_a, Qh);
  gemm_proj_kernel<<<gProj, 32, 0, stream>>>(f1h, wh[1], bk_a, Kh);
  gemm_proj_kernel<<<gProj, 32, 0, stream>>>(f1h, wh[2], bv_a, Vh);
  attn_kernel<<<dim3(NT / 16, NB), 32, 0, stream>>>(Qh, Kh, Vh, f1f, f1h);
  // branch B: Q from updated f1, K/V from f2, gated update on f2
  gemm_proj_kernel<<<gProj, 32, 0, stream>>>(f1h, wh[3], bq_b, Qh);
  gemm_proj_kernel<<<gProj, 32, 0, stream>>>(f2h, wh[4], bk_b, Kh);
  gemm_proj_kernel<<<gProj, 32, 0, stream>>>(f2h, wh[5], bv_b, Vh);
  attn_kernel<<<dim3(NT / 16, NB), 32, 0, stream>>>(Qh, Kh, Vh, f2f, f2h);

  // concat along time -> [B, F, 1024]
  concat_kernel<<<blocks((size_t)NB * NF * NTW, 256), 256, 0, stream>>>(f1h, f2h, x0h);

  // TCN layers (dilations 1,2,4); TDM-staged slabs; last layer emits pooling partials
  dim3 gTcn(NTW / 16, NB);
  tcn_layer_kernel<<<gTcn, 32, 0, stream>>>(x0h, NF,   cwh[0], conv_b[0], rwh[0], res_b[0], 1, x1h, nullptr);
  tcn_layer_kernel<<<gTcn, 32, 0, stream>>>(x1h, NHID, cwh[1], conv_b[1], rwh[1], res_b[1], 2, x2h, nullptr);
  tcn_layer_kernel<<<gTcn, 32, 0, stream>>>(x2h, NHID, cwh[2], conv_b[2], rwh[2], res_b[2], 4, nullptr, Partial);

  pool_kernel<<<blocks((size_t)NB * NHID, 256), 256, 0, stream>>>(Partial, pooled);
  fc_kernel<<<1, 128, 0, stream>>>(pooled, fc_w, fc_b, out);

  // MMD on updated f1/f2
  sqnorm_kernel<<<blocks((size_t)NB * NT, 256), 256, 0, stream>>>(f1f, sx1);
  sqnorm_kernel<<<blocks((size_t)NB * NT, 256), 256, 0, stream>>>(f2f, sx2);
  dim3 gGram(NT, NB / 16, NB / 16);
  gram_kernel<<<gGram, 32, 0, stream>>>(f1h, f1h, Gxx);
  gram_kernel<<<gGram, 32, 0, stream>>>(f2h, f2h, Gyy);
  gram_kernel<<<gGram, 32, 0, stream>>>(f1h, f2h, Gxy);
  mmd_partial_kernel<<<NT, 256, 0, stream>>>(Gxx, Gyy, Gxy, sx1, sx2, mmdp);
  mmd_final_kernel<<<1, 256, 0, stream>>>(mmdp, out + NB * NOUT);
}
